// MultiheadAttention_80874234184083
// MI455X (gfx1250) — matrix operand order repaired and checked by reading, not compiled, not run on hardware
//
#include <hip/hip_runtime.h>
#include <hip/hip_bf16.h>

#define EMBED 1024
#define NHEAD 16
#define HDIM  64
#define BATCH 4
#define SEQ   2048
#define MTOT  (BATCH * SEQ)   // 8192 rows

typedef __attribute__((ext_vector_type(16))) __bf16 v16bf;
typedef __attribute__((ext_vector_type(8)))  float  v8f;

union FragBF {
    v16bf v;
    uint4 q[2];
    unsigned short u[16];
};

__device__ __forceinline__ unsigned short f32_to_bf16_rn(float f) {
    unsigned int u = __builtin_bit_cast(unsigned int, f);
    u += 0x7fffu + ((u >> 16) & 1u);          // round-to-nearest-even
    return (unsigned short)(u >> 16);
}

// Async copy 16B global -> LDS (CDNA5 GLOBAL_LOAD_ASYNC_TO_LDS_B128).
// lds_off = byte offset within the workgroup's LDS allocation,
// gaddr   = 64-bit global address. Tracked by ASYNCcnt.
__device__ __forceinline__ void async_g2l_b128(unsigned lds_off,
                                               const void* gaddr) {
    asm volatile("global_load_async_to_lds_b128 %0, %1, off"
                 :
                 : "v"(lds_off), "v"((unsigned long long)gaddr)
                 : "memory");
}
__device__ __forceinline__ void wait_asynccnt0() {
    asm volatile("s_wait_asynccnt 0x0" ::: "memory");
}

// ---------------------------------------------------------------------------
// f32 -> bf16 bulk convert
// ---------------------------------------------------------------------------
__global__ void mha_cvt_bf16(const float* __restrict__ in,
                             unsigned short* __restrict__ out, int n) {
    int i = blockIdx.x * blockDim.x + threadIdx.x;
    int stride = gridDim.x * blockDim.x;
    for (; i < n; i += stride) out[i] = f32_to_bf16_rn(in[i]);
}

// ---------------------------------------------------------------------------
// Tiled WMMA GEMM:  out[m,n] = (sum_k X[m,k] * W[n,k] + bias[n]) * scale
// X: [M,1024] bf16 row-major, W: [1024,1024] bf16 row-major (rows = out feat)
// SCATTER_QKV: store bf16 into [B,H,S,D]; else store f32 row-major [M,E].
// ---------------------------------------------------------------------------
#define BM  128
#define BN  128
#define BKK 32
#define LDT (BKK + 8)   // 40 halves = 80B row stride (16B aligned)
#define ABYTES (BM * LDT * 2)
#define BBYTES (BN * LDT * 2)

template <bool SCATTER_QKV>
__global__ __launch_bounds__(256) void mha_gemm_wmma(
    const unsigned short* __restrict__ X, const unsigned short* __restrict__ W,
    const float* __restrict__ bias, void* __restrict__ Out, float scale) {
    // single carved LDS block so async-copy byte offsets are compile-time
    __shared__ __align__(16) unsigned char gsm[ABYTES + BBYTES];
    unsigned short(*As)[LDT] = (unsigned short(*)[LDT])gsm;
    unsigned short(*Bs)[LDT] = (unsigned short(*)[LDT])(gsm + ABYTES);

    const int tid  = threadIdx.x;
    const int lane = tid & 31;
    const int wave = tid >> 5;
    const int wm   = wave >> 1;   // 0..3 (rows of 32)
    const int wn   = wave & 1;    // 0..1 (cols of 64)
    const int m0   = blockIdx.x * BM;
    const int n0   = blockIdx.y * BN;
    const int srow = tid >> 1;          // 0..127
    const int skh  = (tid & 1) * 16;    // 0 or 16 halves
    const int half = lane >> 4;
    const int l16  = lane & 15;
    const unsigned srow_off = (unsigned)srow * (LDT * 2) + (unsigned)skh * 2;

    v8f acc[2][4];
#pragma unroll
    for (int mt = 0; mt < 2; ++mt)
#pragma unroll
        for (int nt = 0; nt < 4; ++nt)
#pragma unroll
            for (int i = 0; i < 8; ++i) acc[mt][nt][i] = 0.0f;

    for (int k0 = 0; k0 < EMBED; k0 += BKK) {
        __syncthreads();
        {   // async global -> LDS staging (no VGPR round-trip)
            const unsigned short* sa = X + (size_t)(m0 + srow) * EMBED + k0 + skh;
            async_g2l_b128(srow_off, sa);
            async_g2l_b128(srow_off + 16, sa + 8);
            const unsigned short* sb = W + (size_t)(n0 + srow) * EMBED + k0 + skh;
            async_g2l_b128(ABYTES + srow_off, sb);
            async_g2l_b128(ABYTES + srow_off + 16, sb + 8);
            if (k0 + BKK < EMBED) {      // global_prefetch_b8 of next k-tile
                __builtin_prefetch(sa + BKK, 0, 0);
                __builtin_prefetch(sb + BKK, 0, 0);
            }
        }
        wait_asynccnt0();
        __syncthreads();

        // hoist all fragments of this k-step so ds_loads batch under one wait
        FragBF a[2];
#pragma unroll
        for (int mt = 0; mt < 2; ++mt) {
            const unsigned short* p =
                &As[wm * 32 + mt * 16 + l16][half * 8];
            a[mt].q[0] = *(const uint4*)p;          // k = base .. base+7
            a[mt].q[1] = *(const uint4*)(p + 16);   // k = base+16 .. base+23
        }
        FragBF b[4];
#pragma unroll
        for (int nt = 0; nt < 4; ++nt) {
            const unsigned short* p =
                &Bs[wn * 64 + nt * 16 + l16][half * 8];
            b[nt].q[0] = *(const uint4*)p;
            b[nt].q[1] = *(const uint4*)(p + 16);
        }
#pragma unroll
        for (int nt = 0; nt < 4; ++nt)
#pragma unroll
            for (int mt = 0; mt < 2; ++mt)
                acc[mt][nt] = __builtin_amdgcn_wmma_f32_16x16x32_bf16(
                    false, a[mt].v, false, b[nt].v, (short)0, acc[mt][nt],
                    false, false);
    }

#pragma unroll
    for (int mt = 0; mt < 2; ++mt)
#pragma unroll
        for (int nt = 0; nt < 4; ++nt)
#pragma unroll
            for (int r = 0; r < 8; ++r) {
                int m = m0 + wm * 32 + mt * 16 + r + (half << 3);
                int n = n0 + wn * 64 + nt * 16 + l16;
                float v = (acc[mt][nt][r] + bias[n]) * scale;
                if (SCATTER_QKV) {
                    unsigned short* o = (unsigned short*)Out;
                    int b_ = m >> 11, s = m & (SEQ - 1);
                    int h = n >> 6, d = n & (HDIM - 1);
                    o[(((size_t)b_ * NHEAD + h) * SEQ + s) * HDIM + d] =
                        f32_to_bf16_rn(v);
                } else {
                    ((float*)Out)[(size_t)m * EMBED + n] = v;
                }
            }
}

// ---------------------------------------------------------------------------
// Flash attention: per (b,h) x 128-query tile. 8 waves x 16 query rows.
// kv tiles of 64, online softmax in f32, bf16 WMMA for QK^T and PV.
// Row sums are computed by a WMMA against an all-ones B fragment (result is
// replicated across N columns -> no cross-lane reduction needed for l).
// Q/K staging uses async global->LDS copies; V needs a transpose so it keeps
// the register path. Q/K/V in [B,H,S,D] bf16; output attn in [B,S,H*D] bf16.
// ---------------------------------------------------------------------------
#define LDA 72                        // 72 halves = 144B row stride
#define QP_BYTES (128 * LDA * 2)      // Qs / Ps region (offset 0)
#define K_BYTES  (64 * LDA * 2)
#define K_BASE   QP_BYTES
#define VT_BASE  (QP_BYTES + K_BYTES)

__global__ __launch_bounds__(256) void mha_attn_wmma(
    const unsigned short* __restrict__ Q, const unsigned short* __restrict__ K,
    const unsigned short* __restrict__ V, unsigned short* __restrict__ O) {
    __shared__ __align__(16) unsigned char smem[QP_BYTES + 2 * K_BYTES];
    unsigned short(*QPs)[LDA] = (unsigned short(*)[LDA])smem;
    unsigned short(*Ks)[LDA]  = (unsigned short(*)[LDA])(smem + K_BASE);
    unsigned short(*Vts)[LDA] = (unsigned short(*)[LDA])(smem + VT_BASE);

    const int tid  = threadIdx.x;
    const int lane = tid & 31;
    const int wave = tid >> 5;
    const int qt   = blockIdx.x;      // query tile (0..15)
    const int bh   = blockIdx.y;      // 0..63
    const int q0   = qt * 128;
    const size_t base = (size_t)bh * SEQ * HDIM;
    const int half = lane >> 4;       // fragment half
    const int l16  = lane & 15;

    // constant all-ones bf16 B fragment (1.0bf16 == 0x3F80)
    FragBF ones;
#pragma unroll
    for (int i = 0; i < 16; ++i) ones.u[i] = 0x3F80u;

    // ---- stage Q tile (async): 256 threads cover 128 rows x 64 d ----
    {
        int row = tid >> 1, dh = (tid & 1) * 32;
        const unsigned short* src = Q + base + (size_t)(q0 + row) * HDIM + dh;
        unsigned qoff = (unsigned)row * (LDA * 2) + (unsigned)dh * 2;
        async_g2l_b128(qoff, src);
        async_g2l_b128(qoff + 16, src + 8);
        async_g2l_b128(qoff + 32, src + 16);
        async_g2l_b128(qoff + 48, src + 24);
    }
    wait_asynccnt0();
    __syncthreads();

    FragBF aq[2];
#pragma unroll
    for (int c = 0; c < 2; ++c) {
        const unsigned short* p = &QPs[wave * 16 + l16][c * 32 + half * 8];
        aq[c].q[0] = *(const uint4*)p;
        aq[c].q[1] = *(const uint4*)(p + 16);
    }

    float mrow[8];
    v8f of[4], lf;   // O accumulator + row-sum accumulator (WMMA ones trick)
#pragma unroll
    for (int r = 0; r < 8; ++r) { mrow[r] = -1e30f; lf[r] = 0.0f; }
#pragma unroll
    for (int nt = 0; nt < 4; ++nt)
#pragma unroll
        for (int r = 0; r < 8; ++r) of[nt][r] = 0.0f;

    const int nkt = 2 * qt + 2;   // causal: kv up to q0+127
    for (int kt = 0; kt < nkt; ++kt) {
        __syncthreads();
        {   // stage K [kv][d] (async) and V transposed [d][kv] (register path)
            int row = tid >> 2, dh = (tid & 3) * 16;
            const size_t g = base + (size_t)(kt * 64 + row) * HDIM + dh;
            unsigned koff =
                K_BASE + (unsigned)row * (LDA * 2) + (unsigned)dh * 2;
            async_g2l_b128(koff, K + g);
            async_g2l_b128(koff + 16, K + g + 8);
            FragBF vv;
            vv.q[0] = *(const uint4*)(V + g);
            vv.q[1] = *(const uint4*)(V + g + 8);
#pragma unroll
            for (int i = 0; i < 16; ++i) Vts[dh + i][row] = vv.u[i];
        }
        wait_asynccnt0();
        __syncthreads();

        // ---- S = Q K^T  (16 q-rows x 64 kv); hoist all B frags ----
        FragBF bk[4][2];
#pragma unroll
        for (int t = 0; t < 4; ++t)
#pragma unroll
            for (int c = 0; c < 2; ++c) {
                const unsigned short* p = &Ks[t * 16 + l16][c * 32 + half * 8];
                bk[t][c].q[0] = *(const uint4*)p;
                bk[t][c].q[1] = *(const uint4*)(p + 16);
            }
        v8f sf[4];
#pragma unroll
        for (int t = 0; t < 4; ++t) {
#pragma unroll
            for (int i = 0; i < 8; ++i) sf[t][i] = 0.0f;
            sf[t] = __builtin_amdgcn_wmma_f32_16x16x32_bf16(
                false, aq[0].v, false, bk[t][0].v, (short)0, sf[t], false,
                false);
            sf[t] = __builtin_amdgcn_wmma_f32_16x16x32_bf16(
                false, aq[1].v, false, bk[t][1].v, (short)0, sf[t], false,
                false);
        }

        // ---- causal mask (only tiles touching the diagonal) ----
        if (kt >= 2 * qt) {
#pragma unroll
            for (int t = 0; t < 4; ++t)
#pragma unroll
                for (int r = 0; r < 8; ++r) {
                    int qg = q0 + wave * 16 + r + (half << 3);
                    int kg = kt * 64 + t * 16 + l16;
                    if (kg > qg) sf[t][r] = -1e30f;
                }
        }

        // ---- online softmax: row max (shuffle), rescale ----
        float rm[8];
#pragma unroll
        for (int r = 0; r < 8; ++r) {
            rm[r] = sf[0][r];
#pragma unroll
            for (int t = 1; t < 4; ++t) rm[r] = fmaxf(rm[r], sf[t][r]);
        }
#pragma unroll
        for (int r = 0; r < 8; ++r) {
            rm[r] = fmaxf(rm[r], __shfl_xor(rm[r], 1, 32));
            rm[r] = fmaxf(rm[r], __shfl_xor(rm[r], 2, 32));
            rm[r] = fmaxf(rm[r], __shfl_xor(rm[r], 4, 32));
            rm[r] = fmaxf(rm[r], __shfl_xor(rm[r], 8, 32));
        }
#pragma unroll
        for (int r = 0; r < 8; ++r) {
            float nm    = fmaxf(mrow[r], rm[r]);
            float alpha = __expf(mrow[r] - nm);
            mrow[r]     = nm;
            lf[r] *= alpha;
#pragma unroll
            for (int nt = 0; nt < 4; ++nt) of[nt][r] *= alpha;
        }
        // P = exp(S - m) -> per-wave LDS region (A-layout source)
#pragma unroll
        for (int t = 0; t < 4; ++t)
#pragma unroll
            for (int r = 0; r < 8; ++r) {
                float pv = __expf(sf[t][r] - mrow[r]);
                QPs[(wave << 4) + r + (half << 3)][t * 16 + l16] =
                    f32_to_bf16_rn(pv);
            }

        // ---- O += P V ; l += P * ones (row sums via WMMA) ----
#pragma unroll
        for (int c = 0; c < 2; ++c) {
            FragBF ap;
            const unsigned short* p =
                &QPs[(wave << 4) + l16][c * 32 + half * 8];
            ap.q[0] = *(const uint4*)p;
            ap.q[1] = *(const uint4*)(p + 16);
            FragBF bv[4];
#pragma unroll
            for (int nt = 0; nt < 4; ++nt) {
                const unsigned short* pb =
                    &Vts[nt * 16 + l16][c * 32 + half * 8];
                bv[nt].q[0] = *(const uint4*)pb;
                bv[nt].q[1] = *(const uint4*)(pb + 16);
            }
#pragma unroll
            for (int nt = 0; nt < 4; ++nt)
                of[nt] = __builtin_amdgcn_wmma_f32_16x16x32_bf16(
                    false, ap.v, false, bv[nt].v, (short)0, of[nt], false,
                    false);
            lf = __builtin_amdgcn_wmma_f32_16x16x32_bf16(
                false, ap.v, false, ones.v, (short)0, lf, false, false);
        }
    }

    // ---- finalize: O / l, store bf16 to [B, S, H*D] ----
    const int b = bh >> 4, h = bh & 15;
#pragma unroll
    for (int r = 0; r < 8; ++r) {
        float inv = 1.0f / lf[r];
        int s = q0 + wave * 16 + r + (half << 3);
#pragma unroll
        for (int nt = 0; nt < 4; ++nt) {
            int d = nt * 16 + l16;
            O[((size_t)(b * SEQ + s)) * EMBED + h * HDIM + d] =
                f32_to_bf16_rn(of[nt][r] * inv);
        }
    }
}

// ---------------------------------------------------------------------------
extern "C" void kernel_launch(void* const* d_in, const int* in_sizes, int n_in,
                              void* d_out, int out_size, void* d_ws,
                              size_t ws_size, hipStream_t stream) {
    (void)in_sizes; (void)n_in; (void)out_size; (void)ws_size;
    const float* query = (const float*)d_in[0];
    const float* Wq = (const float*)d_in[1];
    const float* bq = (const float*)d_in[2];
    const float* Wk = (const float*)d_in[3];
    const float* bk = (const float*)d_in[4];
    const float* Wv = (const float*)d_in[5];
    const float* bv = (const float*)d_in[6];
    const float* Wo = (const float*)d_in[7];
    const float* bo = (const float*)d_in[8];

    unsigned char* ws = (unsigned char*)d_ws;
    size_t off = 0;
    unsigned short* Xb  = (unsigned short*)(ws + off); off += (size_t)MTOT * EMBED * 2;
    unsigned short* Wqb = (unsigned short*)(ws + off); off += (size_t)EMBED * EMBED * 2;
    unsigned short* Wkb = (unsigned short*)(ws + off); off += (size_t)EMBED * EMBED * 2;
    unsigned short* Wvb = (unsigned short*)(ws + off); off += (size_t)EMBED * EMBED * 2;
    unsigned short* Wob = (unsigned short*)(ws + off); off += (size_t)EMBED * EMBED * 2;
    unsigned short* Qb  = (unsigned short*)(ws + off); off += (size_t)MTOT * EMBED * 2;
    unsigned short* Kb  = (unsigned short*)(ws + off); off += (size_t)MTOT * EMBED * 2;
    unsigned short* Vb  = (unsigned short*)(ws + off); off += (size_t)MTOT * EMBED * 2;
    unsigned short* Ab  = (unsigned short*)(ws + off); off += (size_t)MTOT * EMBED * 2;

    const int nX = MTOT * EMBED;      // 8388608
    const int nW = EMBED * EMBED;     // 1048576
    mha_cvt_bf16<<<2048, 256, 0, stream>>>(query, Xb, nX);
    mha_cvt_bf16<<<1024, 256, 0, stream>>>(Wq, Wqb, nW);
    mha_cvt_bf16<<<1024, 256, 0, stream>>>(Wk, Wkb, nW);
    mha_cvt_bf16<<<1024, 256, 0, stream>>>(Wv, Wvb, nW);
    mha_cvt_bf16<<<1024, 256, 0, stream>>>(Wo, Wob, nW);

    dim3 gg(MTOT / BM, EMBED / BN);   // 64 x 8
    const float scaling = 0.125f;     // HEAD_DIM^-0.5
    mha_gemm_wmma<true><<<gg, 256, 0, stream>>>(Xb, Wqb, bq, Qb, scaling);
    mha_gemm_wmma<true><<<gg, 256, 0, stream>>>(Xb, Wkb, bk, Kb, 1.0f);
    mha_gemm_wmma<true><<<gg, 256, 0, stream>>>(Xb, Wvb, bv, Vb, 1.0f);

    mha_attn_wmma<<<dim3(SEQ / 128, BATCH * NHEAD), 256, 0, stream>>>(Qb, Kb,
                                                                      Vb, Ab);

    mha_gemm_wmma<false><<<gg, 256, 0, stream>>>(Ab, Wob, bo, d_out, 1.0f);
}
